// TransformerBlock_25159918420147
// MI455X (gfx1250) — matrix operand order repaired and checked by reading, not compiled, not run on hardware
//
#include <hip/hip_runtime.h>
#include <stdint.h>

#define GRP   4
#define NH    8
#define DHEAD 64
#define GDIM  256
#define NCHAN 1024
#define INNER 2048
#define NB    2
#define SEQ   1024
#define EPSLN 1e-5f

typedef __attribute__((ext_vector_type(16))) __bf16         v16bf;
typedef __attribute__((ext_vector_type(8)))  float          v8f;
typedef __attribute__((ext_vector_type(16))) unsigned short v16u;
typedef __attribute__((ext_vector_type(8)))  unsigned short v8u;
typedef __attribute__((ext_vector_type(4)))  float          f32x4;

__device__ __forceinline__ unsigned short f2bf(float f) {
  unsigned int u = __builtin_bit_cast(unsigned int, f);
  return (unsigned short)((u + 0x7FFFu + ((u >> 16) & 1u)) >> 16);  // RNE
}

__device__ __forceinline__ v8f wmma_bf16(v16bf a, v16bf b, v8f c) {
  return __builtin_amdgcn_wmma_f32_16x16x32_bf16(false, a, false, b, (short)0, c, false, false);
}

// ---- fragment loaders (CDNA5 wave32 VGPR layouts, CDNA5 ISA §7.12.2) ----
// A (16x32 bf16) row-major [M,K]: lane<16 row m0+lane K {kc..kc+7, kc+16..kc+23};
// lane>=16 same rows, K {kc+8..15, kc+24..31}
__device__ __forceinline__ v16bf load_a_bf16(const uint16_t* A, int lda, int m0, int k0) {
  int lane = threadIdx.x & 31;
  const uint16_t* p = A + (size_t)(m0 + (lane & 15)) * lda + k0 + ((lane >> 4) << 3);
  v8u lo = *(const v8u*)p;
  v8u hi = *(const v8u*)(p + 16);
  v16u u;
#pragma unroll
  for (int i = 0; i < 8; ++i) { u[i] = lo[i]; u[i + 8] = hi[i]; }
  return __builtin_bit_cast(v16bf, u);
}

// B (32x16 bf16): element (k,n) stored column-major at B[n*ldn + k].
// lane holds column n0+(lane&15); lanes 0-15 K {0..7,16..23}, lanes 16-31 K {8..15,24..31}.
__device__ __forceinline__ v16bf load_b_cm(const uint16_t* B, int ldn, int k0, int n0) {
  int lane = threadIdx.x & 31;
  const uint16_t* p = B + (size_t)(n0 + (lane & 15)) * ldn + k0 + ((lane >> 4) << 3);
  v8u lo = *(const v8u*)p;
  v8u hi = *(const v8u*)(p + 16);
  v16u u;
#pragma unroll
  for (int i = 0; i < 8; ++i) { u[i] = lo[i]; u[i + 8] = hi[i]; }
  return __builtin_bit_cast(v16bf, u);
}

// ---- one-shot fp32 -> bf16 weight conversion (hoists cvt chains out of GEMM loops) ----
__global__ void cvt_bf16_kernel(const float* __restrict__ in, uint16_t* __restrict__ out) {
  size_t i = ((size_t)blockIdx.x * blockDim.x + threadIdx.x) * 4;
  f32x4 v = *(const f32x4*)(in + i);
  out[i]     = f2bf(v[0]);
  out[i + 1] = f2bf(v[1]);
  out[i + 2] = f2bf(v[2]);
  out[i + 3] = f2bf(v[3]);
}

// ---- GroupLayerNorm: x f32 [b,C,n] -> xn bf16 [b][n][C] (column-major for B operands) ----
__global__ void gln_kernel(const float* __restrict__ x, const float* __restrict__ gamma,
                           const float* __restrict__ beta, uint16_t* __restrict__ out) {
  __shared__ float s1[256], s2[256], smean[GRP], srstd[GRP];
  int b = blockIdx.x / SEQ, pos = blockIdx.x % SEQ;
  int t = threadIdx.x;
  float vals[GRP];
#pragma unroll
  for (int g = 0; g < GRP; ++g)
    vals[g] = x[((size_t)b * NCHAN + g * GDIM + t) * SEQ + pos];
  for (int g = 0; g < GRP; ++g) {
    s1[t] = vals[g]; s2[t] = vals[g] * vals[g];
    __syncthreads();
    for (int off = 128; off > 0; off >>= 1) {
      if (t < off) { s1[t] += s1[t + off]; s2[t] += s2[t + off]; }
      __syncthreads();
    }
    if (t == 0) {
      float mean = s1[0] * (1.0f / GDIM);
      float var  = fmaxf(s2[0] * (1.0f / GDIM) - mean * mean, 0.0f);
      smean[g] = mean;
      srstd[g] = 1.0f / (sqrtf(var) + EPSLN);   // denom is std+eps per reference
    }
    __syncthreads();
  }
#pragma unroll
  for (int g = 0; g < GRP; ++g) {
    float v = (vals[g] - smean[g]) * srstd[g] * gamma[g * GDIM + t] + beta[g * GDIM + t];
    out[((size_t)b * SEQ + pos) * NCHAN + g * GDIM + t] = f2bf(v);
  }
}

// one wave -> 32x64 C block (2 m-tiles share B fragments -> half the B traffic).
// Ping-pong (unroll-by-2) register double buffer: 8 WMMAs per stage over a 12x b128
// refill clause, partial s_wait_loadcnt under the WMMAs. Requires K % 64 == 0.
// Needs ~190 VGPRs: callers use __launch_bounds__(256,1) so RA doesn't spill.
__device__ __forceinline__ void gemm_strip2(const uint16_t* A, int lda, const uint16_t* B, int ldn,
                                            int m0, int n0, int K, v8f acc[2][4]) {
  v16bf a0[2], a1[2], b0[4], b1[4];
#pragma unroll
  for (int mi = 0; mi < 2; ++mi) a0[mi] = load_a_bf16(A, lda, m0 + 16 * mi, 0);
#pragma unroll
  for (int t = 0; t < 4; ++t) b0[t] = load_b_cm(B, ldn, 0, n0 + 16 * t);
#pragma unroll
  for (int mi = 0; mi < 2; ++mi) a1[mi] = load_a_bf16(A, lda, m0 + 16 * mi, 32);
#pragma unroll
  for (int t = 0; t < 4; ++t) b1[t] = load_b_cm(B, ldn, 32, n0 + 16 * t);
  for (int kc = 0; kc < K; kc += 64) {
#pragma unroll
    for (int mi = 0; mi < 2; ++mi)
#pragma unroll
      for (int t = 0; t < 4; ++t) acc[mi][t] = wmma_bf16(a0[mi], b0[t], acc[mi][t]);
    if (kc + 64 < K) {
#pragma unroll
      for (int mi = 0; mi < 2; ++mi) a0[mi] = load_a_bf16(A, lda, m0 + 16 * mi, kc + 64);
#pragma unroll
      for (int t = 0; t < 4; ++t) b0[t] = load_b_cm(B, ldn, kc + 64, n0 + 16 * t);
    }
#pragma unroll
    for (int mi = 0; mi < 2; ++mi)
#pragma unroll
      for (int t = 0; t < 4; ++t) acc[mi][t] = wmma_bf16(a1[mi], b1[t], acc[mi][t]);
    if (kc + 96 < K) {
#pragma unroll
      for (int mi = 0; mi < 2; ++mi) a1[mi] = load_a_bf16(A, lda, m0 + 16 * mi, kc + 96);
#pragma unroll
      for (int t = 0; t < 4; ++t) b1[t] = load_b_cm(B, ldn, kc + 96, n0 + 16 * t);
    }
  }
}

// ---- Q projection: Wq[2048,1024] x xn -> q bf16 [b][gh][n][dh], scaled by dh^-0.5 ----
__global__ void __launch_bounds__(256, 1)
gemm_q_kernel(const uint16_t* __restrict__ Wq, const uint16_t* __restrict__ xn,
              uint16_t* __restrict__ qb) {
  int wid = (blockIdx.x * blockDim.x + threadIdx.x) >> 5;   // 2048 waves
  int b = wid >> 10, rem = wid & 1023;                      // 64 m-strips x 16 n-strips
  int m0 = (rem >> 4) << 5, n0 = (rem & 15) << 6;
  v8f acc[2][4] = {};
  gemm_strip2(Wq, NCHAN, xn + (size_t)b * SEQ * NCHAN, NCHAN, m0, n0, NCHAN, acc);
  int lane = threadIdx.x & 31, hi = lane >> 4;
#pragma unroll
  for (int mi = 0; mi < 2; ++mi)
#pragma unroll
    for (int t = 0; t < 4; ++t)
#pragma unroll
      for (int r = 0; r < 8; ++r) {
        int m = m0 + 16 * mi + r + (hi << 3);
        int i = n0 + 16 * t + (lane & 15);
        qb[(((size_t)b * 32 + (m >> 6)) * SEQ + i) * DHEAD + (m & 63)] =
            f2bf(acc[mi][t][r] * 0.125f);
      }
}

// ---- KV projection: k -> [b][gh][n][dh] (Kt operand), v -> [b][gh][dh][n] (channel-major) ----
__global__ void __launch_bounds__(256, 1)
gemm_kv_kernel(const uint16_t* __restrict__ Wkv, const uint16_t* __restrict__ xn,
               uint16_t* __restrict__ kb, uint16_t* __restrict__ vb) {
  int wid = (blockIdx.x * blockDim.x + threadIdx.x) >> 5;   // 4096 waves
  int b = wid >> 11, rem = wid & 2047;                      // 128 m-strips x 16 n-strips
  int m0 = (rem >> 4) << 5, n0 = (rem & 15) << 6;
  v8f acc[2][4] = {};
  gemm_strip2(Wkv, NCHAN, xn + (size_t)b * SEQ * NCHAN, NCHAN, m0, n0, NCHAN, acc);
  int lane = threadIdx.x & 31, hi = lane >> 4;
#pragma unroll
  for (int mi = 0; mi < 2; ++mi)
#pragma unroll
    for (int t = 0; t < 4; ++t)
#pragma unroll
      for (int r = 0; r < 8; ++r) {
        int m = m0 + 16 * mi + r + (hi << 3);
        int i = n0 + 16 * t + (lane & 15);
        unsigned short v = f2bf(acc[mi][t][r]);
        if (m < INNER)
          kb[(((size_t)b * 32 + (m >> 6)) * SEQ + i) * DHEAD + (m & 63)] = v;
        else {
          int m2 = m - INNER;
          vb[(((size_t)b * 32 + (m2 >> 6)) * DHEAD + (m2 & 63)) * SEQ + i] = v;
        }
      }
}

// Q/K fragment triple for one 32-wide k-chunk of the concatenated head dim
struct QK { v16bf a, b0, b1; };
__device__ __forceinline__ QK load_qk(const uint16_t* qb, const uint16_t* kb, int base, int h,
                                      int kc, int i0, int j0) {
  int gp = kc >> 6, dl = kc & 63;
  const uint16_t* qp = qb + ((size_t)(base + gp * NH + h)) * SEQ * DHEAD;
  const uint16_t* kp = kb + ((size_t)(base + gp * NH + h)) * SEQ * DHEAD;
  QK r;
  r.a  = load_a_bf16(qp, DHEAD, i0, dl);
  r.b0 = load_b_cm(kp, DHEAD, dl, j0);
  r.b1 = load_b_cm(kp, DHEAD, dl, j0 + 16);
  return r;
}

// ---- flash attention, group-cumsum folded into concatenated head dim d_eff=(g+1)*64 ----
__global__ void attn_kernel(const uint16_t* __restrict__ qb, const uint16_t* __restrict__ kb,
                            const uint16_t* __restrict__ vb, uint16_t* __restrict__ ob) {
  __shared__ uint16_t plds[8][16 * 32];     // per-wave P tile (bf16)
  int lane = threadIdx.x & 31, hi = lane >> 4;
  int widb = threadIdx.x >> 5;
  int wid = blockIdx.x * (blockDim.x >> 5) + widb;   // 4096 waves: (b,g,h,iblock)
  int ib = wid & 63, tmp = wid >> 6;
  int h = tmp & 7;  tmp >>= 3;
  int g = tmp & 3;  int b = tmp >> 2;
  int i0 = ib << 4;
  int deff = (g + 1) * DHEAD;     // always a multiple of 64
  int base = b * 32;
  uint16_t* pl = &plds[widb][0];
  const uint16_t* vp = vb + ((size_t)(base + g * NH + h)) * DHEAD * SEQ;

  v8f o[4] = {};
  float mst[8], lst[8];
#pragma unroll
  for (int r = 0; r < 8; ++r) { mst[r] = -1e30f; lst[r] = 0.0f; }

  for (int j0 = 0; j0 <= i0 + 15; j0 += 32) {
    v8f s[2] = {};
    QK c0 = load_qk(qb, kb, base, h, 0, i0, j0);
    QK c1 = load_qk(qb, kb, base, h, 32, i0, j0);
    for (int kc = 0; kc < deff; kc += 64) {
      s[0] = wmma_bf16(c0.a, c0.b0, s[0]);
      s[1] = wmma_bf16(c0.a, c0.b1, s[1]);
      if (kc + 64 < deff) c0 = load_qk(qb, kb, base, h, kc + 64, i0, j0);
      s[0] = wmma_bf16(c1.a, c1.b0, s[0]);
      s[1] = wmma_bf16(c1.a, c1.b1, s[1]);
      if (kc + 96 < deff) c1 = load_qk(qb, kb, base, h, kc + 96, i0, j0);
    }
    v16bf vf[4];
#pragma unroll
    for (int t4 = 0; t4 < 4; ++t4) vf[t4] = load_b_cm(vp, SEQ, j0, t4 * 16);
#pragma unroll
    for (int t = 0; t < 2; ++t)
#pragma unroll
      for (int r = 0; r < 8; ++r) {
        int i = i0 + r + (hi << 3);
        int j = j0 + 16 * t + (lane & 15);
        if (j > i) s[t][r] = -1e30f;
      }
#pragma unroll
    for (int r = 0; r < 8; ++r) {
      float v = fmaxf(s[0][r], s[1][r]);
#pragma unroll
      for (int off = 1; off < 16; off <<= 1) v = fmaxf(v, __shfl_xor(v, off, 32));
      float nm = fmaxf(mst[r], v);
      float sc = expf(mst[r] - nm);
      mst[r] = nm;
      s[0][r] = expf(s[0][r] - nm);
      s[1][r] = expf(s[1][r] - nm);
      float rs = s[0][r] + s[1][r];
#pragma unroll
      for (int off = 1; off < 16; off <<= 1) rs += __shfl_xor(rs, off, 32);
      lst[r] = lst[r] * sc + rs;
#pragma unroll
      for (int t4 = 0; t4 < 4; ++t4) o[t4][r] *= sc;
    }
#pragma unroll
    for (int t = 0; t < 2; ++t)
#pragma unroll
      for (int r = 0; r < 8; ++r)
        pl[(r + (hi << 3)) * 32 + 16 * t + (lane & 15)] = f2bf(s[t][r]);
    v16bf pa = load_a_bf16(pl, 32, 0, 0);
#pragma unroll
    for (int t4 = 0; t4 < 4; ++t4)
      o[t4] = wmma_bf16(pa, vf[t4], o[t4]);
  }
#pragma unroll
  for (int t4 = 0; t4 < 4; ++t4)
#pragma unroll
    for (int r = 0; r < 8; ++r) {
      int i = i0 + r + (hi << 3);
      int d = t4 * 16 + (lane & 15);
      ob[((size_t)b * SEQ + i) * INNER + (g * NH + h) * DHEAD + d] = f2bf(o[t4][r] / lst[r]);
    }
}

// ---- Wo: [1024,2048] x o + bo + x(residual) -> x1 f32 [b,C,n] ----
__global__ void __launch_bounds__(256, 1)
gemm_o_kernel(const uint16_t* __restrict__ Wo, const uint16_t* __restrict__ ob,
              const float* __restrict__ bo, const float* __restrict__ xres,
              float* __restrict__ x1) {
  int wid = (blockIdx.x * blockDim.x + threadIdx.x) >> 5;   // 1024 waves
  int b = wid >> 9, rem = wid & 511;                        // 32 m-strips x 16 n-strips
  int m0 = (rem >> 4) << 5, n0 = (rem & 15) << 6;
  v8f acc[2][4] = {};
  gemm_strip2(Wo, INNER, ob + (size_t)b * SEQ * INNER, INNER, m0, n0, INNER, acc);
  int lane = threadIdx.x & 31, hi = lane >> 4;
#pragma unroll
  for (int mi = 0; mi < 2; ++mi)
#pragma unroll
    for (int t = 0; t < 4; ++t)
#pragma unroll
      for (int r = 0; r < 8; ++r) {
        int m = m0 + 16 * mi + r + (hi << 3);
        int i = n0 + 16 * t + (lane & 15);
        size_t idx = ((size_t)b * NCHAN + m) * SEQ + i;
        x1[idx] = acc[mi][t][r] + bo[m] + xres[idx];
      }
}

// ---- FFN1: per-group Wf1[g][1024,256] x xn2_g, +bias, exact GELU -> h bf16 [b][g][n][1024] ----
__global__ void __launch_bounds__(256, 1)
gemm_f1_kernel(const uint16_t* __restrict__ Wf1, const uint16_t* __restrict__ xn,
               const float* __restrict__ bf1, uint16_t* __restrict__ hb) {
  int wid = (blockIdx.x * blockDim.x + threadIdx.x) >> 5;   // 4096 waves
  int bg = wid >> 9, rem = wid & 511;                       // 32 m-strips x 16 n-strips
  int b = bg >> 2, g = bg & 3;
  int m0 = (rem >> 4) << 5, n0 = (rem & 15) << 6;
  v8f acc[2][4] = {};
  gemm_strip2(Wf1 + (size_t)g * 1024 * GDIM, GDIM,
              xn + (size_t)b * SEQ * NCHAN + g * GDIM, NCHAN, m0, n0, GDIM, acc);
  int lane = threadIdx.x & 31, hi = lane >> 4;
#pragma unroll
  for (int mi = 0; mi < 2; ++mi)
#pragma unroll
    for (int t = 0; t < 4; ++t)
#pragma unroll
      for (int r = 0; r < 8; ++r) {
        int m = m0 + 16 * mi + r + (hi << 3);
        int i = n0 + 16 * t + (lane & 15);
        float v = acc[mi][t][r] + bf1[g * 1024 + m];
        v = 0.5f * v * (1.0f + erff(v * 0.70710678118f));   // exact GELU
        hb[(((size_t)b * GRP + g) * SEQ + i) * 1024 + m] = f2bf(v);
      }
}

// ---- FFN2: per-group Wf2[g][256,1024] x h_g, + bf2 + x1 -> out f32 [b,C,n] ----
__global__ void __launch_bounds__(256, 1)
gemm_f2_kernel(const uint16_t* __restrict__ Wf2, const uint16_t* __restrict__ hb,
               const float* __restrict__ bf2, const float* __restrict__ x1,
               float* __restrict__ out) {
  int wid = (blockIdx.x * blockDim.x + threadIdx.x) >> 5;   // 1024 waves
  int bg = wid >> 7, rem = wid & 127;                       // 8 m-strips x 16 n-strips
  int b = bg >> 2, g = bg & 3;
  int m0 = (rem >> 4) << 5, n0 = (rem & 15) << 6;
  v8f acc[2][4] = {};
  gemm_strip2(Wf2 + (size_t)g * GDIM * 1024, 1024,
              hb + ((size_t)b * GRP + g) * SEQ * 1024, 1024, m0, n0, 1024, acc);
  int lane = threadIdx.x & 31, hi = lane >> 4;
#pragma unroll
  for (int mi = 0; mi < 2; ++mi)
#pragma unroll
    for (int t = 0; t < 4; ++t)
#pragma unroll
      for (int r = 0; r < 8; ++r) {
        int ch = g * GDIM + m0 + 16 * mi + r + (hi << 3);
        int i = n0 + 16 * t + (lane & 15);
        size_t idx = ((size_t)b * NCHAN + ch) * SEQ + i;
        out[idx] = acc[mi][t][r] + bf2[ch] + x1[idx];
      }
}

extern "C" void kernel_launch(void* const* d_in, const int* in_sizes, int n_in,
                              void* d_out, int out_size, void* d_ws, size_t ws_size,
                              hipStream_t stream) {
  const float* x   = (const float*)d_in[0];
  const float* g1  = (const float*)d_in[1];
  const float* b1  = (const float*)d_in[2];
  const float* Wq  = (const float*)d_in[3];
  const float* Wkv = (const float*)d_in[4];
  const float* Wo  = (const float*)d_in[5];
  const float* bo  = (const float*)d_in[6];
  const float* g2  = (const float*)d_in[7];
  const float* b2  = (const float*)d_in[8];
  const float* Wf1 = (const float*)d_in[9];
  const float* bf1 = (const float*)d_in[10];
  const float* Wf2 = (const float*)d_in[11];
  const float* bf2 = (const float*)d_in[12];
  float* out = (float*)d_out;

  char* ws = (char*)d_ws;
  uint16_t* xn  = (uint16_t*)(ws);                      //  4 MB  [b][n][C]   bf16
  uint16_t* qb  = (uint16_t*)(ws + (4u  << 20));        //  8 MB  [b][gh][n][dh]
  uint16_t* kb  = (uint16_t*)(ws + (12u << 20));        //  8 MB  [b][gh][n][dh]
  uint16_t* vb  = (uint16_t*)(ws + (20u << 20));        //  8 MB  [b][gh][dh][n]
  uint16_t* ob  = (uint16_t*)(ws + (28u << 20));        //  8 MB  [b][n][inner]
  float*    x1  = (float*)   (ws + (36u << 20));        //  8 MB  [b][C][n] f32
  uint16_t* hb  = (uint16_t*)(ws + (44u << 20));        // 16 MB  [b][g][n][hid/g]
  uint16_t* wqb = (uint16_t*)(ws + (60u << 20));        //  4 MB  Wq bf16
  uint16_t* wkvb= (uint16_t*)(ws + (64u << 20));        //  8 MB  Wkv bf16
  uint16_t* wob = (uint16_t*)(ws + (72u << 20));        //  4 MB  Wo bf16
  uint16_t* wf1b= (uint16_t*)(ws + (76u << 20));        //  2 MB  Wf1 bf16
  uint16_t* wf2b= (uint16_t*)(ws + (78u << 20));        //  2 MB  Wf2 bf16
  (void)in_sizes; (void)n_in; (void)out_size; (void)ws_size;

  cvt_bf16_kernel<<<2048, 256, 0, stream>>>(Wq,  wqb);   // 2M elems
  cvt_bf16_kernel<<<4096, 256, 0, stream>>>(Wkv, wkvb);  // 4M elems
  cvt_bf16_kernel<<<2048, 256, 0, stream>>>(Wo,  wob);   // 2M elems
  cvt_bf16_kernel<<<1024, 256, 0, stream>>>(Wf1, wf1b);  // 1M elems
  cvt_bf16_kernel<<<1024, 256, 0, stream>>>(Wf2, wf2b);  // 1M elems

  gln_kernel    <<<NB * SEQ, 256, 0, stream>>>(x, g1, b1, xn);
  gemm_q_kernel <<<256,  256, 0, stream>>>(wqb, xn, qb);
  gemm_kv_kernel<<<512,  256, 0, stream>>>(wkvb, xn, kb, vb);
  attn_kernel   <<<512,  256, 0, stream>>>(qb, kb, vb, ob);
  gemm_o_kernel <<<128,  256, 0, stream>>>(wob, ob, bo, x, x1);
  gln_kernel    <<<NB * SEQ, 256, 0, stream>>>(x1, g2, b2, xn);
  gemm_f1_kernel<<<512,  256, 0, stream>>>(wf1b, xn, bf1, hb);
  gemm_f2_kernel<<<128,  256, 0, stream>>>(wf2b, hb, bf2, x1, out);
}
